// EdgeExtractionGraphConvolutional_7524782702855
// MI455X (gfx1250) — matrix operand order repaired and checked by reading, not compiled, not run on hardware
//
#include <hip/hip_runtime.h>
#include <hip/hip_bf16.h>
#include <stdint.h>

typedef __attribute__((ext_vector_type(16))) _Float16 v16h;
typedef __attribute__((ext_vector_type(8)))  float   v8f;

#define XKS   168   // LDS activation row stride in halves (160 max K + 8 pad -> conflict-free)
#define WAVES 2     // waves per block (blockDim = 64)
#define MT    2     // 16-row tiles per wave (32 rows/wave, shares B fragments)
#define MR    (MT * 16)

// f16 blocked-weight region offsets (in halves), layout [Kpad/32][Npad][32]
#define NW1_OFF   0u
#define NW2_OFF   12288u
#define E0_OFF    20480u
#define E1_OFF    40960u
#define E2_OFF    57344u
#define E3_OFF    65536u
#define E4_OFF    73728u
#define E5_OFF    90112u
#define H0_OFF    94208u
#define HP_STRIDE 81920u
#define W_TOTAL_HALVES 421888u

#define FIXP 16777216.0f   // 2^24 fixed point for deterministic segment-sum

__device__ __forceinline__ void wave_lds_fence() {
  // per-wave LDS producer->consumer fence (LDS ops are in-order per wave)
  asm volatile("s_wait_dscnt 0" ::: "memory");
}

#define ACT_SILU  0
#define ACT_LEAKY 1
#define ACT_NONE  2

template<int ACT>
__device__ __forceinline__ float act_apply(float x) {
  if (ACT == ACT_SILU)  return x * __builtin_amdgcn_rcpf(1.f + __expf(-x));
  if (ACT == ACT_LEAKY) return (x > 0.f) ? x : 0.01f * x;
  return x;
}

// One dense layer for an MR-row tile: Y[MR,Npad] = act(X[MR,Kpad] @ W + b)
// X in LDS f16 (row stride XKS). W blocked f16 [Kpad/32][Npad][32].
// MT accumulator tiles share each B fragment (halves weight-fetch traffic).
template<int ACT>
__device__ void layerM(const _Float16* xs,
                       const _Float16* __restrict__ Wc,
                       const float* __restrict__ bias,
                       int Kpad, int Npad, int dout,
                       _Float16* ys,
                       float* __restrict__ gout, int gld,
                       const int* rowids, int validRows)
{
  const int lane = (int)(threadIdx.x & 31u);
  const int half = lane >> 4;       // 0: lanes 0-15, 1: lanes 16-31
  const int mh   = lane & 15;       // A row (within tile) / C column
  for (int n0 = 0; n0 < Npad; n0 += 16) {
    const int n = n0 + mh;
    const float bv = (n < dout) ? bias[n] : 0.f;
    v8f acc[MT];
#pragma unroll
    for (int t = 0; t < MT; ++t) acc[t] = (v8f){bv, bv, bv, bv, bv, bv, bv, bv};
    for (int kb = 0; kb < Kpad; kb += 32) {
      v16h b;
      uint32_t* bp = reinterpret_cast<uint32_t*>(&b);
      const _Float16* wb = Wc + ((size_t)(kb >> 5) * (size_t)Npad + (size_t)n) * 32u + (half << 4);
#pragma unroll
      for (int v = 0; v < 8; ++v) {
        // B k order used here: lane n = col, K = half*16 + 2v (+0,+1 packed) -> contiguous in blocked W
        bp[v] = *reinterpret_cast<const uint32_t*>(wb + (v << 1));
      }
#pragma unroll
      for (int t = 0; t < MT; ++t) {
        v16h a;
        uint32_t* ap = reinterpret_cast<uint32_t*>(&a);
        const _Float16* xrow = xs + (t * 16 + mh) * XKS + kb;
#pragma unroll
        for (int v = 0; v < 8; ++v) {
          // A k order here (contiguous, not ISA table order): K = half*16 + 2v (+0,+1 packed), matching B
          const int kloc = (half << 4) + (v << 1);
          ap[v] = *reinterpret_cast<const uint32_t*>(xrow + kloc);
        }
        acc[t] = __builtin_amdgcn_wmma_f32_16x16x32_f16(false, a, false, b, (short)0, acc[t], false, false);
      }
    }
#pragma unroll
    for (int t = 0; t < MT; ++t) {
#pragma unroll
      for (int r = 0; r < 8; ++r) {
        const float x = act_apply<ACT>(acc[t][r]);
        const int row = t * 16 + r + (half << 3);   // C layout: lanes16-31 hold M = r+8
        if (ys) ys[row * XKS + n0 + mh] = (_Float16)x;
        if (gout != nullptr) {
          if (row < validRows && n < dout)
            gout[(size_t)rowids[row] * (size_t)gld + (size_t)n] = x;
        }
      }
    }
  }
}

// ---------------- stage kernels ----------------

__global__ void k_zero(unsigned long long* agg, int n, int* counts) {
  int i = blockIdx.x * blockDim.x + threadIdx.x;
  if (i < n) agg[i] = 0ull;
  if (i < 4) counts[i] = 0;
}

// fp32 [din,dout] -> f16 blocked [Kpad/32][Npad][32], zero padded
__global__ void k_convw(const float* __restrict__ src, _Float16* __restrict__ dst,
                        int din, int dout, int Kpad, int Npad) {
  int t = blockIdx.x * blockDim.x + threadIdx.x;
  int total = Kpad * Npad;
  if (t >= total) return;
  int kloc = t & 31;
  int rest = t >> 5;
  int n  = rest % Npad;
  int kb = rest / Npad;
  int k = (kb << 5) + kloc;
  float v = (k < din && n < dout) ? src[(size_t)k * dout + n] : 0.f;
  dst[t] = (_Float16)v;
}

// deterministic segment-sum of ec over dst via 2^24 fixed-point int64 atomics
__global__ void k_scatter(const float* __restrict__ er, const float* __restrict__ ea,
                          const int* __restrict__ ei, int E,
                          unsigned long long* agg) {
  int i = blockIdx.x * blockDim.x + threadIdx.x;
  if (i >= E * 17) return;
  int e = i / 17, c = i % 17;
  float v = (c < 8) ? er[(size_t)e * 8 + c] : ea[(size_t)e * 9 + (c - 8)];
  long long q = llrintf(v * FIXP);
  atomicAdd(&agg[(size_t)ei[E + e] * 17 + c], (unsigned long long)q);
}

__global__ void k_bucket(const int* __restrict__ ei, const int* __restrict__ at, int E,
                         int* counts, int* lists) {
  int e = blockIdx.x * blockDim.x + threadIdx.x;
  if (e >= E) return;
  int p = at[ei[e]] * 2 + at[ei[E + e]];
  int pos = atomicAdd(&counts[p], 1);
  lists[(size_t)p * E + pos] = e;
}

// node update: [nf(64) | agg(17) | 0pad] -> 128 silu -> 64
__global__ void k_node(const float* __restrict__ nf, const unsigned long long* __restrict__ agg,
                       int N, const _Float16* __restrict__ W,
                       const float* __restrict__ b1, const float* __restrict__ b2,
                       float* __restrict__ nodes_upd) {
  __shared__ _Float16 sx[WAVES][2][MR * XKS];
  __shared__ int srow[WAVES][MR];
  const int wave = threadIdx.x >> 5, lane = threadIdx.x & 31;
  const int tile = blockIdx.x * WAVES + wave;
  const int r0 = tile * MR;
  _Float16* x0 = &sx[wave][0][0];
  _Float16* x1 = &sx[wave][1][0];
  for (int t = lane; t < MR * 96; t += 32) {
    int row = t / 96, col = t % 96;
    int node = r0 + row;
    float v = 0.f;
    if (node < N) {
      if (col < 64)      v = nf[(size_t)node * 64 + col];
      else if (col < 81) v = (float)(long long)agg[(size_t)node * 17 + (col - 64)] * (1.0f / FIXP);
    }
    x0[row * XKS + col] = (_Float16)v;
    if (col == 0) srow[wave][row] = (node < N) ? node : 0;
  }
  int valid = N - r0; valid = valid < 0 ? 0 : (valid > MR ? MR : valid);
  wave_lds_fence();
  layerM<ACT_SILU>(x0, W + NW1_OFF, b1, 96, 128, 128, x1, nullptr, 0, nullptr, 0);
  wave_lds_fence();
  layerM<ACT_NONE>(x1, W + NW2_OFF, b2, 128, 64, 64, nullptr, nodes_upd, 64, srow[wave], valid);
}

// edge update: [nf[src](64) | nf[dst](64) | er(8) | ea(9) | 0pad] -> 6-layer MLP -> edges_upd[E,17]
__global__ void k_edge(const float* __restrict__ nf, const float* __restrict__ er,
                       const float* __restrict__ ea, const int* __restrict__ ei,
                       int E, const _Float16* __restrict__ W,
                       const float* __restrict__ b0, const float* __restrict__ b1,
                       const float* __restrict__ b2, const float* __restrict__ b3,
                       const float* __restrict__ b4, const float* __restrict__ b5,
                       float* __restrict__ edges_upd) {
  __shared__ _Float16 sx[WAVES][2][MR * XKS];
  __shared__ int srow[WAVES][MR];
  const int wave = threadIdx.x >> 5, lane = threadIdx.x & 31;
  const int tile = blockIdx.x * WAVES + wave;
  const int e0 = tile * MR;
  _Float16* x0 = &sx[wave][0][0];
  _Float16* x1 = &sx[wave][1][0];
  for (int t = lane; t < MR * 160; t += 32) {
    int row = t / 160, col = t % 160;
    int e = e0 + row;
    float v = 0.f;
    if (e < E) {
      if (col < 64)       v = nf[(size_t)ei[e] * 64 + col];
      else if (col < 128) v = nf[(size_t)ei[E + e] * 64 + (col - 64)];
      else if (col < 136) v = er[(size_t)e * 8 + (col - 128)];
      else if (col < 145) v = ea[(size_t)e * 9 + (col - 136)];
    }
    x0[row * XKS + col] = (_Float16)v;
    if (col == 0) srow[wave][row] = (e < E) ? e : 0;
  }
  int valid = E - e0; valid = valid < 0 ? 0 : (valid > MR ? MR : valid);
  wave_lds_fence();
  layerM<ACT_SILU >(x0, W + E0_OFF, b0, 160, 128, 128, x1, nullptr, 0, nullptr, 0);
  wave_lds_fence();
  layerM<ACT_SILU >(x1, W + E1_OFF, b1, 128, 128, 128, x0, nullptr, 0, nullptr, 0);
  wave_lds_fence();
  layerM<ACT_SILU >(x0, W + E2_OFF, b2, 128, 64, 64,   x1, nullptr, 0, nullptr, 0);
  wave_lds_fence();
  layerM<ACT_LEAKY>(x1, W + E3_OFF, b3, 64, 128, 128,  x0, nullptr, 0, nullptr, 0);
  wave_lds_fence();
  layerM<ACT_SILU >(x0, W + E4_OFF, b4, 128, 128, 128, x1, nullptr, 0, nullptr, 0);
  wave_lds_fence();
  layerM<ACT_NONE >(x1, W + E5_OFF, b5, 128, 32, 17, nullptr, edges_upd, 17, srow[wave], valid);
}

// pair-routed head: hx = [nu[src](64) | nu[dst](64) | eu(17) | 0pad] -> 5-layer MLP -> out[e,81]
__global__ void k_head(const float* __restrict__ nu, const float* __restrict__ eu,
                       const int* __restrict__ ei,
                       const int* __restrict__ counts, const int* __restrict__ lists,
                       int E, const _Float16* __restrict__ W,
                       const float* __restrict__ hb0, const float* __restrict__ hb1,
                       const float* __restrict__ hb2, const float* __restrict__ hb3,
                       const float* __restrict__ hb4,
                       float* __restrict__ out) {
  __shared__ _Float16 sx[WAVES][2][MR * XKS];
  __shared__ int srow[WAVES][MR];
  const int wave = threadIdx.x >> 5, lane = threadIdx.x & 31;
  const int p = blockIdx.y;
  const int cnt = counts[p];
  const int tile = blockIdx.x * WAVES + wave;
  const int i0 = tile * MR;
  if (i0 >= cnt) return;   // no workgroup barriers used -> safe per-wave exit
  _Float16* x0 = &sx[wave][0][0];
  _Float16* x1 = &sx[wave][1][0];
  const _Float16* Wp = W + H0_OFF + (size_t)p * HP_STRIDE;
  for (int t = lane; t < MR * 160; t += 32) {
    int row = t / 160, col = t % 160;
    int i = i0 + row;
    float v = 0.f;
    int e = 0;
    if (i < cnt) {
      e = lists[(size_t)p * E + i];
      if (col < 64)       v = nu[(size_t)ei[e] * 64 + col];
      else if (col < 128) v = nu[(size_t)ei[E + e] * 64 + (col - 64)];
      else if (col < 145) v = eu[(size_t)e * 17 + (col - 128)];
    }
    x0[row * XKS + col] = (_Float16)v;
    if (col == 0) srow[wave][row] = e;
  }
  int valid = cnt - i0; valid = valid > MR ? MR : valid;
  wave_lds_fence();
  layerM<ACT_SILU >(x0, Wp + 0u,     hb0 + (size_t)p * 128, 160, 128, 128, x1, nullptr, 0, nullptr, 0);
  wave_lds_fence();
  layerM<ACT_SILU >(x1, Wp + 20480u, hb1 + (size_t)p * 128, 128, 128, 128, x0, nullptr, 0, nullptr, 0);
  wave_lds_fence();
  layerM<ACT_SILU >(x0, Wp + 36864u, hb2 + (size_t)p * 128, 128, 128, 128, x1, nullptr, 0, nullptr, 0);
  wave_lds_fence();
  layerM<ACT_LEAKY>(x1, Wp + 53248u, hb3 + (size_t)p * 128, 128, 128, 128, x0, nullptr, 0, nullptr, 0);
  wave_lds_fence();
  layerM<ACT_NONE >(x0, Wp + 69632u, hb4 + (size_t)p * 81, 128, 96, 81, nullptr, out, 81, srow[wave], valid);
}

// ---------------- host launcher ----------------

struct CvDesc { const float* src; size_t dstOffHalves; int din, dout, Kpad, Npad; };

extern "C" void kernel_launch(void* const* d_in, const int* in_sizes, int n_in,
                              void* d_out, int out_size, void* d_ws, size_t ws_size,
                              hipStream_t stream) {
  const float* nf = (const float*)d_in[0];
  const float* er = (const float*)d_in[1];
  const float* ea = (const float*)d_in[2];
  const int*   ei = (const int*)d_in[3];
  const int*   at = (const int*)d_in[4];

  const int N = in_sizes[0] / 64;
  const int E = in_sizes[1] / 8;

  // Resolve params flattening order (dict insertion order vs jax sorted-key order)
  int iNW1, iNW2, iNB1, iNB2, iEW0, iEB0, iHW0, iHB0;
  if (in_sizes[5] == 81 * 128) {       // insertion order: node{W1,b1,W2,b2}, edge{W,b}, head{W,b}
    iNW1 = 5; iNB1 = 6; iNW2 = 7; iNB2 = 8;
    iEW0 = 9;  iEB0 = 15;
    iHW0 = 21; iHB0 = 26;
  } else {                              // jax sorted keys: edge{W,b}, head{W,b}, node{W1,W2,b1,b2}
    iEW0 = 5;  iEB0 = 11;
    iHW0 = 17; iHB0 = 22;
    iNW1 = 27; iNW2 = 28; iNB1 = 29; iNB2 = 30;
  }
  const float* eW[6]; const float* eb[6];
  for (int i = 0; i < 6; ++i) { eW[i] = (const float*)d_in[iEW0 + i]; eb[i] = (const float*)d_in[iEB0 + i]; }
  const float* hW[5]; const float* hb[5];
  for (int i = 0; i < 5; ++i) { hW[i] = (const float*)d_in[iHW0 + i]; hb[i] = (const float*)d_in[iHB0 + i]; }
  const float* nW1 = (const float*)d_in[iNW1];
  const float* nW2 = (const float*)d_in[iNW2];
  const float* nb1 = (const float*)d_in[iNB1];
  const float* nb2 = (const float*)d_in[iNB2];

  // workspace layout
  char* ws = (char*)d_ws;
  size_t o = 0;
  auto alignup = [](size_t x) { return (x + 255) & ~(size_t)255; };
  const size_t aggOff = o; o = alignup(o + (size_t)N * 17 * 8);
  const size_t nuOff  = o; o = alignup(o + (size_t)N * 64 * 4);
  const size_t euOff  = o; o = alignup(o + (size_t)E * 17 * 4);
  const size_t cntOff = o; o = alignup(o + 4 * sizeof(int));
  const size_t lstOff = o; o = alignup(o + (size_t)4 * E * sizeof(int));
  const size_t wOff   = o; o += (size_t)W_TOTAL_HALVES * 2;
  (void)ws_size;

  unsigned long long* agg = (unsigned long long*)(ws + aggOff);
  float* nodes_upd = (float*)(ws + nuOff);
  float* edges_upd = (float*)(ws + euOff);
  int* counts = (int*)(ws + cntOff);
  int* lists  = (int*)(ws + lstOff);
  _Float16* Wbase = (_Float16*)(ws + wOff);
  float* out = (float*)d_out;

  // 1) zero agg + counts
  {
    int n = N * 17;
    k_zero<<<(n + 255) / 256, 256, 0, stream>>>(agg, n, counts);
  }

  // 2) convert all weights to blocked f16
  CvDesc cv[28]; int nc = 0;
  cv[nc++] = { nW1, NW1_OFF, 81, 128, 96, 128 };
  cv[nc++] = { nW2, NW2_OFF, 128, 64, 128, 64 };
  {
    const int edin[6]  = {145, 128, 128, 64, 128, 128};
    const int edout[6] = {128, 128, 64, 128, 128, 17};
    const int ekp[6]   = {160, 128, 128, 64, 128, 128};
    const int enp[6]   = {128, 128, 64, 128, 128, 32};
    const size_t eoff[6] = {E0_OFF, E1_OFF, E2_OFF, E3_OFF, E4_OFF, E5_OFF};
    for (int l = 0; l < 6; ++l)
      cv[nc++] = { eW[l], eoff[l], edin[l], edout[l], ekp[l], enp[l] };
    const int hdin[5]  = {145, 128, 128, 128, 128};
    const int hdout[5] = {128, 128, 128, 128, 81};
    const int hkp[5]   = {160, 128, 128, 128, 128};
    const int hnp[5]   = {128, 128, 128, 128, 96};
    const size_t hoff[5] = {0, 20480, 36864, 53248, 69632};
    for (int p = 0; p < 4; ++p)
      for (int l = 0; l < 5; ++l)
        cv[nc++] = { hW[l] + (size_t)p * hdin[l] * hdout[l],
                     (size_t)H0_OFF + (size_t)p * HP_STRIDE + hoff[l],
                     hdin[l], hdout[l], hkp[l], hnp[l] };
  }
  for (int i = 0; i < nc; ++i) {
    int total = cv[i].Kpad * cv[i].Npad;
    k_convw<<<(total + 255) / 256, 256, 0, stream>>>(cv[i].src, Wbase + cv[i].dstOffHalves,
                                                     cv[i].din, cv[i].dout, cv[i].Kpad, cv[i].Npad);
  }

  // 3) deterministic segment-sum + 4) pair buckets
  {
    int n = E * 17;
    k_scatter<<<(n + 255) / 256, 256, 0, stream>>>(er, ea, ei, E, agg);
    k_bucket<<<(E + 255) / 256, 256, 0, stream>>>(ei, at, E, counts, lists);
  }

  // 5) node MLP
  {
    int tiles = (N + MR - 1) / MR;
    k_node<<<(tiles + WAVES - 1) / WAVES, WAVES * 32, 0, stream>>>(nf, agg, N, Wbase, nb1, nb2, nodes_upd);
  }

  // 6) edge MLP
  {
    int tiles = (E + MR - 1) / MR;
    k_edge<<<(tiles + WAVES - 1) / WAVES, WAVES * 32, 0, stream>>>(nf, er, ea, ei, E, Wbase,
                                                                  eb[0], eb[1], eb[2], eb[3], eb[4], eb[5],
                                                                  edges_upd);
  }

  // 7) routed heads
  {
    int tiles = (E + MR - 1) / MR;
    dim3 grid((tiles + WAVES - 1) / WAVES, 4);
    k_head<<<grid, WAVES * 32, 0, stream>>>(nodes_upd, edges_upd, ei, counts, lists, E, Wbase,
                                            hb[0], hb[1], hb[2], hb[3], hb[4], out);
  }
}
